// ExpressiveMambaEncoder_52913997086997
// MI455X (gfx1250) — matrix operand order repaired and checked by reading, not compiled, not run on hardware
//
#include <hip/hip_runtime.h>
#include <hip/hip_bf16.h>
#include <cstdint>
#include <cstddef>

typedef __attribute__((ext_vector_type(16))) _Float16 v16h;
typedef __attribute__((ext_vector_type(8)))  _Float16 v8h;
typedef __attribute__((ext_vector_type(8)))  float    v8f;
typedef int v4i_vs __attribute__((vector_size(16)));   // matches builtin param type

#define BB 4
#define CCH 512
#define TT 2048
#define DI 1024     // D_INNER
#define DS 16       // D_STATE
#define DTR 32      // DT_RANK
#define NDEPTH 4
#define BT (BB*TT)  // 8192

#if __has_builtin(__builtin_amdgcn_global_load_async_to_lds_b128)
#define HAVE_ASYNC 1
#else
#define HAVE_ASYNC 0
#endif

__device__ __forceinline__ float silu_f(float x) { return x * (1.0f / (1.0f + __expf(-x))); }

__device__ __forceinline__ void wait_async0() {
#if HAVE_ASYNC
#if __has_builtin(__builtin_amdgcn_s_wait_asynccnt)
    __builtin_amdgcn_s_wait_asynccnt(0);
#else
    asm volatile("s_wait_asynccnt 0x0" ::: "memory");
#endif
#endif
}

#if HAVE_ASYNC
__device__ __forceinline__ void async_b128(const _Float16* g, _Float16* l) {
    __builtin_amdgcn_global_load_async_to_lds_b128(
        (__attribute__((address_space(1))) v4i_vs*)(uintptr_t)g,
        (__attribute__((address_space(3))) v4i_vs*)(unsigned)(uintptr_t)l,
        0, 0);
}
#endif

// ---------------------------------------------------------------------------
// Large WMMA GEMM: C[M,N](f32,ldc) = A[M,K](f16,lda) x W[N,K]^T(f16,ldw)
// 128x128 block tile, 256 threads (8 waves, 2x4), wave = 64x32 sub-tile,
// K-step 64, double-buffered LDS fed by GLOBAL_LOAD_ASYNC_TO_LDS_B128.
// Requires M%128==0, N%128==0, K%64==0.
// ---------------------------------------------------------------------------
__global__ __launch_bounds__(256)
void wmma_gemm_big(const _Float16* __restrict__ A, int lda,
                   const _Float16* __restrict__ W, int ldw,
                   float* __restrict__ C, int ldc,
                   int M, int N, int K, int accum)
{
    __shared__ __align__(32) _Float16 As[2][128 * 64];
    __shared__ __align__(32) _Float16 Ws[2][128 * 64];
    const int tid   = threadIdx.x;
    const int wave  = tid >> 5;
    const int lane  = tid & 31;
    const int lhalf = lane & 15;
    const int lsel  = lane >> 4;
    const int row0  = blockIdx.x * 128;
    const int col0  = blockIdx.y * 128;
    const int wm    = (wave >> 2) * 64;   // 0 / 64
    const int wn    = (wave & 3) * 32;    // 0..96
    const int lrow  = tid >> 1;           // 0..127
    const int lcol  = (tid & 1) * 32;     // 0 / 32 halves

    v8f acc[4][2] = {};

    const _Float16* ga0 = A + (size_t)(row0 + lrow) * lda + lcol;
    const _Float16* gw0 = W + (size_t)(col0 + lrow) * ldw + lcol;

    auto stage = [&](int buf, int k0) {
        const _Float16* ga = ga0 + k0;
        const _Float16* gw = gw0 + k0;
        _Float16* as = &As[buf][lrow * 64 + lcol];
        _Float16* ws = &Ws[buf][lrow * 64 + lcol];
#if HAVE_ASYNC
        #pragma unroll
        for (int q = 0; q < 4; ++q) {
            async_b128(ga + q * 8, as + q * 8);
            async_b128(gw + q * 8, ws + q * 8);
        }
#else
        #pragma unroll
        for (int q = 0; q < 4; ++q) {
            *(float4*)(as + q * 8) = *(const float4*)(ga + q * 8);
            *(float4*)(ws + q * 8) = *(const float4*)(gw + q * 8);
        }
#endif
    };

    auto compute = [&](int buf) {
        #pragma unroll
        for (int kk = 0; kk < 64; kk += 32) {
            v16h bfr[2];
            #pragma unroll
            for (int ni = 0; ni < 2; ++ni)
                bfr[ni] = *(const v16h*)&Ws[buf][(wn + ni * 16 + lhalf) * 64 + kk + lsel * 16];
            #pragma unroll
            for (int mi = 0; mi < 4; ++mi) {
                const _Float16* p = &As[buf][(wm + mi * 16 + lhalf) * 64 + kk];
                v8h lo = *(const v8h*)(p + lsel * 16);
                v8h hi = *(const v8h*)(p + lsel * 16 + 8);
                v16h f;
                #pragma unroll
                for (int e = 0; e < 8; ++e) { f[e] = lo[e]; f[e + 8] = hi[e]; }
                #pragma unroll
                for (int ni = 0; ni < 2; ++ni)
                    acc[mi][ni] = __builtin_amdgcn_wmma_f32_16x16x32_f16(
                        false, f, false, bfr[ni], (short)0, acc[mi][ni], false, false);
            }
        }
    };

    stage(0, 0);
    wait_async0();
    __syncthreads();
    int cur = 0;
    for (int k0 = 0; k0 < K; k0 += 64) {
        const int nk = k0 + 64;
        if (nk < K) stage(cur ^ 1, nk);      // prefetch next tile while computing
        compute(cur);
        if (nk < K) wait_async0();
        __syncthreads();
        cur ^= 1;
    }

    #pragma unroll
    for (int mi = 0; mi < 4; ++mi)
        #pragma unroll
        for (int ni = 0; ni < 2; ++ni) {
            int r = row0 + wm + mi * 16 + lsel * 8;
            int c = col0 + wn + ni * 16 + lhalf;
            float* o = C + (size_t)r * ldc + c;
            #pragma unroll
            for (int v = 0; v < 8; ++v) {
                float val = acc[mi][ni][v];
                if (accum) o[(size_t)v * ldc] += val;
                else       o[(size_t)v * ldc] = val;
            }
        }
}

// ---------------------------------------------------------------------------
// Small WMMA GEMM (64x64 tile, 128 threads) for odd shapes: N=64 or K=32.
// ---------------------------------------------------------------------------
__global__ __launch_bounds__(128)
void wmma_gemm_f16(const _Float16* __restrict__ A, int lda,
                   const _Float16* __restrict__ W, int ldw,
                   float* __restrict__ C, int ldc,
                   int M, int N, int K, int accum)
{
    __shared__ __align__(32) _Float16 As[64 * 32];
    __shared__ __align__(32) _Float16 Ws[64 * 32];
    const int tid   = threadIdx.x;
    const int wave  = tid >> 5;
    const int lane  = tid & 31;
    const int lhalf = lane & 15;
    const int lsel  = lane >> 4;
    const int row0  = blockIdx.x * 64;
    const int col0  = blockIdx.y * 64;
    const int wm    = (wave >> 1) * 32;
    const int wn    = (wave & 1) * 32;
    const int lrow  = tid >> 1;
    const int lcol  = (tid & 1) * 16;

    v8f acc[2][2] = {};

    for (int k0 = 0; k0 < K; k0 += 32) {
        const _Float16* ga = A + (size_t)(row0 + lrow) * lda + k0 + lcol;
        const _Float16* gw = W + (size_t)(col0 + lrow) * ldw + k0 + lcol;
        *(float4*)&As[lrow * 32 + lcol]     = *(const float4*)(ga);
        *(float4*)&As[lrow * 32 + lcol + 8] = *(const float4*)(ga + 8);
        *(float4*)&Ws[lrow * 32 + lcol]     = *(const float4*)(gw);
        *(float4*)&Ws[lrow * 32 + lcol + 8] = *(const float4*)(gw + 8);
        if (k0 + 32 < K) {
            __builtin_prefetch(ga + 32, 0, 1);
            __builtin_prefetch(gw + 32, 0, 1);
        }
        __syncthreads();

        v16h bfr[2];
        #pragma unroll
        for (int ni = 0; ni < 2; ++ni)
            bfr[ni] = *(const v16h*)&Ws[(wn + ni * 16 + lhalf) * 32 + lsel * 16];

        #pragma unroll
        for (int mi = 0; mi < 2; ++mi) {
            const _Float16* p = &As[(wm + mi * 16 + lhalf) * 32];
            v8h lo = *(const v8h*)(p + lsel * 16);
            v8h hi = *(const v8h*)(p + lsel * 16 + 8);
            v16h f;
            #pragma unroll
            for (int e = 0; e < 8; ++e) { f[e] = lo[e]; f[e + 8] = hi[e]; }
            #pragma unroll
            for (int ni = 0; ni < 2; ++ni)
                acc[mi][ni] = __builtin_amdgcn_wmma_f32_16x16x32_f16(
                    false, f, false, bfr[ni], (short)0, acc[mi][ni], false, false);
        }
        __syncthreads();
    }

    #pragma unroll
    for (int mi = 0; mi < 2; ++mi)
        #pragma unroll
        for (int ni = 0; ni < 2; ++ni) {
            int r = row0 + wm + mi * 16 + lsel * 8;
            int c = col0 + wn + ni * 16 + lhalf;
            float* o = C + (size_t)r * ldc + c;
            #pragma unroll
            for (int v = 0; v < 8; ++v) {
                float val = acc[mi][ni][v];
                if (accum) o[(size_t)v * ldc] += val;
                else       o[(size_t)v * ldc] = val;
            }
        }
}

// ---------------------------------------------------------------------------
// Front stack: grouped conv1d (K=3, 4ch/group) + GroupNorm + SiLU, gsum accum.
// ---------------------------------------------------------------------------
__global__ __launch_bounds__(1024)
void front_conv_gn(const float* __restrict__ hin, float* __restrict__ hout,
                   float* __restrict__ gsum,
                   const float* __restrict__ cw, const float* __restrict__ cb,
                   const float* __restrict__ gw, const float* __restrict__ gb,
                   int first)
{
    const int b = blockIdx.x >> 2;
    const int g = blockIdx.x & 3;
    const int NEL = 128 * TT;
    __shared__ float ssum[1024];
    __shared__ float ssq[1024];
    float lsum = 0.f, lsq = 0.f;

    for (int idx = threadIdx.x; idx < NEL; idx += 1024) {
        int cr = idx >> 11;
        int t  = idx & (TT - 1);
        int c  = g * 128 + cr;
        const float* wrow = cw + c * 12;
        int ibase = (c >> 2) << 2;
        float acc = cb[c];
        #pragma unroll
        for (int j = 0; j < 4; ++j) {
            const float* in = hin + ((size_t)b * CCH + ibase + j) * TT;
            float x0 = (t > 0)      ? in[t - 1] : 0.f;
            float x1 = in[t];
            float x2 = (t < TT - 1) ? in[t + 1] : 0.f;
            acc += wrow[j * 3 + 0] * x0 + wrow[j * 3 + 1] * x1 + wrow[j * 3 + 2] * x2;
        }
        hout[((size_t)b * CCH + c) * TT + t] = acc;
        lsum += acc; lsq += acc * acc;
    }
    ssum[threadIdx.x] = lsum; ssq[threadIdx.x] = lsq;
    __syncthreads();
    for (int s = 512; s > 0; s >>= 1) {
        if (threadIdx.x < s) { ssum[threadIdx.x] += ssum[threadIdx.x + s]; ssq[threadIdx.x] += ssq[threadIdx.x + s]; }
        __syncthreads();
    }
    const float inv = 1.0f / (float)NEL;
    float mu   = ssum[0] * inv;
    float var  = ssq[0] * inv - mu * mu;
    float rstd = rsqrtf(var + 1e-5f);

    for (int idx = threadIdx.x; idx < NEL; idx += 1024) {
        int cr = idx >> 11;
        int t  = idx & (TT - 1);
        int c  = g * 128 + cr;
        size_t o = ((size_t)b * CCH + c) * TT + t;
        float v  = hout[o];
        float xn = (v - mu) * rstd * gw[c] + gb[c];
        float s  = silu_f(xn);
        hout[o] = s;
        if (first) gsum[o] = s; else gsum[o] += s;
    }
}

__global__ void build_gsumT(const float* __restrict__ gsum, _Float16* __restrict__ outh, size_t n)
{
    size_t i = (size_t)blockIdx.x * blockDim.x + threadIdx.x;
    if (i >= n) return;
    int c = i & (CCH - 1);
    size_t bt = i >> 9;
    int t = bt & (TT - 1);
    int b = bt >> 11;
    outh[i] = (_Float16)gsum[((size_t)b * CCH + c) * TT + t];
}

__global__ void mamba_conv(const float* __restrict__ xz, const float* __restrict__ w,
                           const float* __restrict__ bias,
                           float* __restrict__ xc32, _Float16* __restrict__ xc16, size_t n)
{
    size_t i = (size_t)blockIdx.x * blockDim.x + threadIdx.x;
    if (i >= n) return;
    int dch = i & (DI - 1);
    size_t bt = i >> 10;
    int t = bt & (TT - 1);
    int b = bt >> 11;
    float acc = bias[dch];
    #pragma unroll
    for (int k = 0; k < 4; ++k) {
        int tt = t + k - 3;
        if (tt >= 0)
            acc += w[dch * 4 + k] * xz[((size_t)(b * TT + tt)) * (2 * DI) + dch];
    }
    float s = silu_f(acc);
    xc32[i] = s;
    xc16[i] = (_Float16)s;
}

__global__ void f32_to_f16_k(const float* __restrict__ in, _Float16* __restrict__ out, size_t n)
{
    size_t i = (size_t)blockIdx.x * blockDim.x + threadIdx.x;
    if (i < n) out[i] = (_Float16)in[i];
}

__global__ void fill0_k(float* __restrict__ p, size_t n)
{
    size_t i = (size_t)blockIdx.x * blockDim.x + threadIdx.x;
    if (i < n) p[i] = 0.f;
}

__global__ void dt_softplus(float* __restrict__ dt, const float* __restrict__ bias, size_t n)
{
    size_t i = (size_t)blockIdx.x * blockDim.x + threadIdx.x;
    if (i >= n) return;
    float v = dt[i] + bias[i & (DI - 1)];
    dt[i] = (v > 20.f) ? v : log1pf(__expf(v));
}

// Selective scan: one thread per (b,d), 16-state recurrence in VGPRs.
// B_t/C_t double-buffered through LDS -> one barrier per timestep.
__global__ __launch_bounds__(256)
void mamba_scan(const float* __restrict__ dt, const float* __restrict__ xc,
                const float* __restrict__ xdbl, const float* __restrict__ A_log,
                float* __restrict__ ys)
{
    const int b = blockIdx.y;
    const int d = blockIdx.x * 256 + threadIdx.x;
    float A[DS], h[DS];
    #pragma unroll
    for (int n = 0; n < DS; ++n) { A[n] = -__expf(A_log[d * DS + n]); h[n] = 0.f; }
    __shared__ float sB[2][DS];
    __shared__ float sC[2][DS];
    for (int t = 0; t < TT; ++t) {
        const int sl = t & 1;
        size_t bt = (size_t)b * TT + t;
        if (threadIdx.x < 2 * DS) {
            float v = xdbl[bt * 64 + DTR + threadIdx.x];
            if (threadIdx.x < DS) sB[sl][threadIdx.x] = v;
            else                  sC[sl][threadIdx.x - DS] = v;
        }
        __syncthreads();
        float dtv = dt[bt * DI + d];
        float du  = dtv * xc[bt * DI + d];
        float y = 0.f;
        #pragma unroll
        for (int n = 0; n < DS; ++n) {
            h[n] = h[n] * __expf(dtv * A[n]) + du * sB[sl][n];
            y += h[n] * sC[sl][n];
        }
        ys[bt * DI + d] = y;
    }
}

__global__ void ygate_k(const float* __restrict__ ys, const float* __restrict__ xc,
                        const float* __restrict__ xz, const float* __restrict__ Dp,
                        _Float16* __restrict__ yg, size_t n)
{
    size_t i = (size_t)blockIdx.x * blockDim.x + threadIdx.x;
    if (i >= n) return;
    int dch = i & (DI - 1);
    size_t bt = i >> 10;
    float z = xz[bt * (2 * DI) + DI + dch];
    float y = (ys[i] + xc[i] * Dp[dch]) * silu_f(z);
    yg[i] = (_Float16)y;
}

__global__ void pack_combined(const float* __restrict__ gsum, const float* __restrict__ mf,
                              float* __restrict__ c32, _Float16* __restrict__ c16, size_t n)
{
    size_t i = (size_t)blockIdx.x * blockDim.x + threadIdx.x;
    if (i >= n) return;
    int ch = i & (2 * CCH - 1);
    size_t bt = i >> 10;
    int t = bt & (TT - 1);
    int b = bt >> 11;
    float v = (ch < CCH) ? gsum[((size_t)b * CCH + ch) * TT + t]
                         : mf[bt * CCH + (ch - CCH)];
    c32[i] = v;
    c16[i] = (_Float16)v;
}

__global__ void gate_mul(float* __restrict__ gp, const float* __restrict__ comb, size_t n)
{
    size_t i = (size_t)blockIdx.x * blockDim.x + threadIdx.x;
    if (i < n) gp[i] = silu_f(gp[i]) * comb[i];
}

__global__ __launch_bounds__(256)
void ln_k(const float* __restrict__ in, const float* __restrict__ w,
          const float* __restrict__ b, _Float16* __restrict__ outh)
{
    __shared__ float s1[256];
    __shared__ float s2[256];
    const size_t base = (size_t)blockIdx.x * 1024;
    float v[4]; float lsum = 0.f, lsq = 0.f;
    #pragma unroll
    for (int j = 0; j < 4; ++j) {
        v[j] = in[base + threadIdx.x * 4 + j];
        lsum += v[j]; lsq += v[j] * v[j];
    }
    s1[threadIdx.x] = lsum; s2[threadIdx.x] = lsq;
    __syncthreads();
    for (int s = 128; s > 0; s >>= 1) {
        if (threadIdx.x < s) { s1[threadIdx.x] += s1[threadIdx.x + s]; s2[threadIdx.x] += s2[threadIdx.x + s]; }
        __syncthreads();
    }
    float mu   = s1[0] * (1.f / 1024.f);
    float var  = s2[0] * (1.f / 1024.f) - mu * mu;
    float rstd = rsqrtf(var + 1e-5f);
    #pragma unroll
    for (int j = 0; j < 4; ++j) {
        int c = threadIdx.x * 4 + j;
        outh[base + c] = (_Float16)((v[j] - mu) * rstd * w[c] + b[c]);
    }
}

__global__ void final_mask(const float* __restrict__ outT, const unsigned char* __restrict__ m,
                           float* __restrict__ out, size_t n)
{
    size_t i = (size_t)blockIdx.x * blockDim.x + threadIdx.x;
    if (i >= n) return;
    int t = i & (TT - 1);
    int c = (i >> 11) & (CCH - 1);
    int b = i >> 20;
    out[i] = m[b * TT + t] ? 0.f : outT[((size_t)b * TT + t) * CCH + c];
}

// ---------------------------------------------------------------------------
extern "C" void kernel_launch(void* const* d_in, const int* in_sizes, int n_in,
                              void* d_out, int out_size, void* d_ws, size_t ws_size,
                              hipStream_t stream)
{
    (void)in_sizes; (void)n_in; (void)out_size; (void)ws_size;

    const float* x        = (const float*)d_in[0];
    const float* conv_w   = (const float*)d_in[2];
    const float* conv_b   = (const float*)d_in[3];
    const float* gn_w     = (const float*)d_in[4];
    const float* gn_b     = (const float*)d_in[5];
    const float* in_projw = (const float*)d_in[6];
    const float* mconv_w  = (const float*)d_in[7];
    const float* mconv_b  = (const float*)d_in[8];
    const float* x_projw  = (const float*)d_in[9];
    const float* dt_projw = (const float*)d_in[10];
    const float* dt_pb    = (const float*)d_in[11];
    const float* A_log    = (const float*)d_in[12];
    const float* Dp       = (const float*)d_in[13];
    const float* out_projw= (const float*)d_in[14];
    const float* gate_w   = (const float*)d_in[15];
    const float* ln_w     = (const float*)d_in[16];
    const float* ln_b     = (const float*)d_in[17];
    const float* proj_w   = (const float*)d_in[18];
    const unsigned char* mmask = (const unsigned char*)d_in[19];
    float* out = (float*)d_out;

    char* ws = (char*)d_ws;
    size_t off = 0;
    auto alloc = [&](size_t bytes) -> char* {
        char* p = ws + off;
        off += (bytes + 255) & ~(size_t)255;
        return p;
    };

    const size_t nBCT  = (size_t)BB * CCH * TT;
    const size_t nBTD  = (size_t)BT * DI;
    const size_t nBT2D = (size_t)BT * 2 * DI;
    const size_t nXDBL = (size_t)BT * 64;

    float*     hA      = (float*)alloc(nBCT * 4);
    float*     hB2     = (float*)alloc(nBCT * 4);
    float*     gsum    = (float*)alloc(nBCT * 4);
    _Float16*  gsumT16 = (_Float16*)alloc(nBCT * 2);
    _Float16*  w16_in  = (_Float16*)alloc((size_t)NDEPTH * 2048 * 512 * 2);
    _Float16*  w16_xp  = (_Float16*)alloc((size_t)NDEPTH * 64 * 1024 * 2);
    _Float16*  w16_dt  = (_Float16*)alloc((size_t)NDEPTH * 1024 * 32 * 2);
    _Float16*  w16_op  = (_Float16*)alloc((size_t)NDEPTH * 512 * 1024 * 2);
    _Float16*  w16_g   = (_Float16*)alloc((size_t)1024 * 512 * 2);
    _Float16*  w16_pj  = (_Float16*)alloc((size_t)512 * 1024 * 2);
    float*     xz      = (float*)alloc(nBT2D * 4);
    float*     xc32    = (float*)alloc(nBTD * 4);
    _Float16*  xc16    = (_Float16*)alloc(nBTD * 2);
    float*     xdbl    = (float*)alloc(nXDBL * 4);
    _Float16*  xdbl16  = (_Float16*)alloc(nXDBL * 2);
    float*     dtb     = (float*)alloc(nBTD * 4);
    float*     ysb     = (float*)alloc(nBTD * 4);
    _Float16*  yg16    = (_Float16*)alloc(nBTD * 2);
    float*     mf      = (float*)alloc((size_t)BT * CCH * 4);
    // late-stage reuse
    float*     combT32 = xz;
    _Float16*  combT16 = xc16;
    float*     gatep   = xc32;
    _Float16*  ln16    = yg16;
    float*     outT    = ysb;

    const int PT = 256;
    auto nb = [](size_t n, int b) { return (unsigned)((n + (size_t)b - 1) / (size_t)b); };

    f32_to_f16_k<<<nb((size_t)NDEPTH*2048*512, PT), PT, 0, stream>>>(in_projw, w16_in, (size_t)NDEPTH*2048*512);
    f32_to_f16_k<<<nb((size_t)NDEPTH*64*1024, PT), PT, 0, stream>>>(x_projw, w16_xp, (size_t)NDEPTH*64*1024);
    f32_to_f16_k<<<nb((size_t)NDEPTH*1024*32, PT), PT, 0, stream>>>(dt_projw, w16_dt, (size_t)NDEPTH*1024*32);
    f32_to_f16_k<<<nb((size_t)NDEPTH*512*1024, PT), PT, 0, stream>>>(out_projw, w16_op, (size_t)NDEPTH*512*1024);
    f32_to_f16_k<<<nb((size_t)1024*512, PT), PT, 0, stream>>>(gate_w, w16_g, (size_t)1024*512);
    f32_to_f16_k<<<nb((size_t)512*1024, PT), PT, 0, stream>>>(proj_w, w16_pj, (size_t)512*1024);

    fill0_k<<<nb((size_t)BT*CCH, PT), PT, 0, stream>>>(mf, (size_t)BT*CCH);

    {
        float* bufs[2] = { hA, hB2 };
        const float* cur = x;
        for (int d = 0; d < NDEPTH; ++d) {
            float* o = bufs[d & 1];
            front_conv_gn<<<BB * 4, 1024, 0, stream>>>(
                cur, o, gsum, conv_w + d*CCH*12, conv_b + d*CCH, gn_w + d*CCH, gn_b + d*CCH, d == 0);
            cur = o;
        }
    }
    build_gsumT<<<nb(nBCT, PT), PT, 0, stream>>>(gsum, gsumT16, nBCT);

    for (int d = 0; d < NDEPTH; ++d) {
        dim3 g1(BT / 128, 2048 / 128);
        wmma_gemm_big<<<g1, 256, 0, stream>>>(gsumT16, 512, w16_in + (size_t)d*2048*512, 512,
                                              xz, 2048, BT, 2048, 512, 0);
        mamba_conv<<<nb(nBTD, PT), PT, 0, stream>>>(xz, mconv_w + d*DI*4, mconv_b + d*DI, xc32, xc16, nBTD);

        dim3 g2(BT / 64, 64 / 64);
        wmma_gemm_f16<<<g2, 128, 0, stream>>>(xc16, 1024, w16_xp + (size_t)d*64*1024, 1024,
                                              xdbl, 64, BT, 64, 1024, 0);
        f32_to_f16_k<<<nb(nXDBL, PT), PT, 0, stream>>>(xdbl, xdbl16, nXDBL);

        dim3 g3(BT / 64, 1024 / 64);
        wmma_gemm_f16<<<g3, 128, 0, stream>>>(xdbl16, 64, w16_dt + (size_t)d*1024*32, 32,
                                              dtb, 1024, BT, 1024, 32, 0);
        dt_softplus<<<nb(nBTD, PT), PT, 0, stream>>>(dtb, dt_pb + d*DI, nBTD);

        mamba_scan<<<dim3(DI / 256, BB), 256, 0, stream>>>(dtb, xc32, xdbl, A_log + (size_t)d*DI*DS, ysb);
        ygate_k<<<nb(nBTD, PT), PT, 0, stream>>>(ysb, xc32, xz, Dp + d*DI, yg16, nBTD);

        dim3 g4(BT / 128, 512 / 128);
        wmma_gemm_big<<<g4, 256, 0, stream>>>(yg16, 1024, w16_op + (size_t)d*512*1024, 1024,
                                              mf, 512, BT, 512, 1024, 1);
    }

    pack_combined<<<nb(nBTD, PT), PT, 0, stream>>>(gsum, mf, combT32, combT16, nBTD);
    dim3 g5(BT / 128, 512 / 128);
    wmma_gemm_big<<<g5, 256, 0, stream>>>(combT16, 1024, w16_g, 512,
                                          gatep, 1024, BT, 512, 512, 0);
    wmma_gemm_big<<<g5, 256, 0, stream>>>(combT16 + 512, 1024, w16_g + (size_t)512*512, 512,
                                          gatep + 512, 1024, BT, 512, 512, 0);
    gate_mul<<<nb(nBTD, PT), PT, 0, stream>>>(gatep, combT32, nBTD);
    ln_k<<<BT, 256, 0, stream>>>(gatep, ln_w, ln_b, ln16);
    wmma_gemm_big<<<g5, 256, 0, stream>>>(ln16, 1024, w16_pj, 1024,
                                          outT, 512, BT, 512, 1024, 0);
    final_mask<<<nb(nBCT, PT), PT, 0, stream>>>(outT, mmask, out, nBCT);
}
